// WindowAttention_5987184411345
// MI455X (gfx1250) — matrix operand order repaired and checked by reading, not compiled, not run on hardware
//
#include <hip/hip_runtime.h>
#include <hip/hip_bf16.h>

// ---------------------------------------------------------------------------
// Swin window attention, MI455X (gfx1250), wave32 + V_WMMA_F32_16X16X32_F16,
// Tensor Data Mover staging of f16 tiles into padded LDS, double-buffered
// software pipeline (TDM + global loads overlap the WMMA stream).
// Pipeline: prep -> qkv GEMM -> attention (S=qk^T, softmax, O=Pv) -> proj GEMM
// ---------------------------------------------------------------------------

typedef __attribute__((ext_vector_type(16))) _Float16 v16h;
typedef __attribute__((ext_vector_type(8)))  float    v8f;
typedef __attribute__((ext_vector_type(4)))  unsigned int u32x4;
typedef __attribute__((ext_vector_type(8)))  int      i32x8;
typedef __attribute__((ext_vector_type(4)))  int      i32x4;

#define DIM       384
#define NUM_HEADS 12
#define HEAD_DIM  32
#define NTOK      49
#define NBATCH    4096
#define NROWS     (NBATCH * NTOK)      // 200704, divisible by 64
#define NQKV      (3 * DIM)            // 1152
#define BM        64
#define BNQ       128                  // N tile (qkv + proj)
#define BK        64
#define NCHUNK    (DIM / BK)           // 6
#define LDPAD     8                    // LDS row padding (f16) -> TDM pad 4 dwords
#define LDST      (BK + LDPAD)         // 72

#if defined(__HIP_DEVICE_COMPILE__) && \
    __has_builtin(__builtin_amdgcn_tensor_load_to_lds) && \
    __has_builtin(__builtin_amdgcn_s_wait_tensorcnt)
#define USE_TDM 1
#else
#define USE_TDM 0
#endif

__device__ __forceinline__ v8f wmma_f16(v16h a, v16h b, v8f c) {
  return __builtin_amdgcn_wmma_f32_16x16x32_f16(
      false, a, false, b, (short)0, c, false, false);
}

// A-fragment (16x32 f16, MxK) from LDS tile [row][k] (k contiguous).
__device__ __forceinline__ v16h load_frag_A(const _Float16* lds, int stride,
                                            int row0, int k0, int lane) {
  v16h a;
  const int m  = lane & 15;
  const int hi = lane >> 4;
  const _Float16* base = lds + (row0 + m) * stride + k0 + hi * 8;
#pragma unroll
  for (int v = 0; v < 8; ++v) {
    const int kk = (v < 4) ? (2 * v) : (2 * v + 8);
    a[2 * v]     = base[kk];
    a[2 * v + 1] = base[kk + 1];
  }
  return a;
}

// B-fragment (32x16 f16, KxN) from LDS tile [n][k] (k contiguous).
__device__ __forceinline__ v16h load_frag_B(const _Float16* lds, int stride,
                                            int n0, int k0, int lane) {
  v16h b;
  const int n  = lane & 15;
  const int hi = lane >> 4;
  const _Float16* base = lds + (n0 + n) * stride + k0 + hi * 8;
#pragma unroll
  for (int v = 0; v < 8; ++v) {
    const int kk = (v < 4) ? (2 * v) : (2 * v + 8);
    b[2 * v]     = base[kk];
    b[2 * v + 1] = base[kk + 1];
  }
  return b;
}

#if USE_TDM
// Low 32 bits of a generic LDS pointer == LDS byte offset (flat aperture map).
__device__ __forceinline__ unsigned lds_off(const void* p) {
  return (unsigned)(unsigned long long)(size_t)p;
}

// TDM 2D tile load, data_size = 2 bytes (f16). All dims in f16 elements.
// Descriptor packing per CDNA5 ISA 8.3/8.4 (D# group 0/1); groups 2/3 unused.
// pad_icode: 0=2,1=4,2=8,3=16,4=32 dwords between pads; pad_acode+1 dwords pad.
__device__ __forceinline__ void tdm_load_2d(
    unsigned lds_addr, const void* gsrc,
    unsigned tile_d0, unsigned tile_d1,
    unsigned tensor_d0, unsigned tensor_d1,
    unsigned stride_d0,
    unsigned pad_icode, unsigned pad_acode, int pad_en) {
  const unsigned long long ga = (unsigned long long)(size_t)gsrc;
  u32x4 g0;
  g0.x = 1u;                                       // count=1, user mode
  g0.y = lds_addr;                                 // lds_addr [63:32]
  g0.z = (unsigned)(ga & 0xffffffffu);             // global_addr lo
  g0.w = (unsigned)((ga >> 32) & 0x1ffffffu)       // global_addr hi (57-bit)
         | (2u << 30);                             // type = 2 ("image")
  unsigned flags = (1u << 16);                     // data_size = 1 -> 2 bytes
  if (pad_en)
    flags |= (1u << 20) | (pad_icode << 22) | (pad_acode << 25);
  i32x8 g1;
  g1[0] = (int)flags;                              // wg_mask=0, flags
  g1[1] = (int)((tensor_d0 & 0xffffu) << 16);      // dim0 lo16 (atomic addr=0)
  g1[2] = (int)(((tensor_d0 >> 16) & 0xffffu) | ((tensor_d1 & 0xffffu) << 16));
  g1[3] = (int)(((tensor_d1 >> 16) & 0xffffu) | ((tile_d0 & 0xffffu) << 16));
  g1[4] = (int)(tile_d1 & 0xffffu);                // tile_dim1, tile_dim2=0
  g1[5] = (int)stride_d0;                          // dim0 stride lo32
  g1[6] = 0;                                       // stride hi / dim1 stride
  g1[7] = 0;
  const i32x4 z4 = {0, 0, 0, 0};
  const i32x8 z8 = {0, 0, 0, 0, 0, 0, 0, 0};
  __builtin_amdgcn_tensor_load_to_lds(g0, g1, z4, z4, z8, 0);
}
#endif  // USE_TDM

// ---------------------------------------------------------------------------
// Kernel 0: prep. Weights -> f16 [n][k]; padded bias matrix [12][64][64]
// (cols >= 49 get -1e30 so softmax masks token padding).
// ---------------------------------------------------------------------------
__global__ __launch_bounds__(256) void prep_kernel(
    const float* __restrict__ qkv_w, const float* __restrict__ proj_w,
    const float* __restrict__ bias_table, const int* __restrict__ rel,
    _Float16* __restrict__ qkvwT, _Float16* __restrict__ projwT,
    float* __restrict__ biasp) {
  const int idx = blockIdx.x * blockDim.x + threadIdx.x;
  const int T1 = NQKV * DIM;             // 442368
  const int T2 = DIM * DIM;              // 147456
  const int T3 = NUM_HEADS * 64 * 64;    // 49152
  if (idx < T1) {
    const int n = idx / DIM, k = idx % DIM;
    qkvwT[idx] = (_Float16)qkv_w[(size_t)k * NQKV + n];
  } else if (idx < T1 + T2) {
    const int j = idx - T1;
    const int n = j / DIM, k = j % DIM;
    projwT[j] = (_Float16)proj_w[(size_t)k * DIM + n];
  } else if (idx < T1 + T2 + T3) {
    const int j = idx - T1 - T2;
    const int h = j / 4096;
    const int m = (j % 4096) / 64;
    const int n = j % 64;
    float v = -1e30f;
    if (m < NTOK && n < NTOK)
      v = bias_table[(size_t)rel[m * NTOK + n] * NUM_HEADS + h];
    biasp[j] = v;
  }
}

// ---------------------------------------------------------------------------
// Kernel 1: QKV GEMM. [200704x384](f32) x [384x1152](f16) -> q/k/v f16
// [b][h][tok][d]. 64x128 tile / 256 threads; wave tile 32x32 (4 WMMA/kstep).
// Double-buffered pipeline: TDM B(c+1) + global A(c+1) issue before compute(c);
// A commit + tensorcnt wait after compute, before the publishing barrier.
// ---------------------------------------------------------------------------
__global__ __launch_bounds__(256) void qkv_gemm_kernel(
    const float* __restrict__ x, const _Float16* __restrict__ wT,
    const float* __restrict__ qkv_b,
    _Float16* __restrict__ qws, _Float16* __restrict__ kws,
    _Float16* __restrict__ vws) {
  __shared__ _Float16 As[2][BM][LDST];
  __shared__ _Float16 Bs[2][BNQ][LDST];
  const int tiles_n = NQKV / BNQ;         // 9
  const int tn = blockIdx.x % tiles_n;
  const int tm = blockIdx.x / tiles_n;
  const int row0 = tm * BM, col0 = tn * BNQ;
  const int tid  = threadIdx.x;
  const int wave = tid >> 5, lane = tid & 31;
  const int wm = (wave & 1) * 32;
  const int wn = (wave >> 1) * 32;
  const int ar = (tid + 0 * 256) >> 4;          // A-stage row (i stride 16 rows)
  const int ac4 = tid & 15;                      // A-stage float4 col

  // ---- prologue: stage chunk 0 into buffer 0 ----
#if USE_TDM
  if (wave == 0)
    tdm_load_2d(lds_off(&Bs[0][0][0]), wT + (size_t)col0 * DIM,
                BK, BNQ, DIM, NQKV, DIM, 4, 3, 1);
#else
#pragma unroll
  for (int i = 0; i < 4; ++i) {
    const int u = tid + i * 256;
    const int n = u >> 3, c8 = u & 7;
    *((uint4*)&Bs[0][n][c8 * 8]) =
        ((const uint4*)(wT + (size_t)(col0 + n) * DIM))[c8];
  }
#endif
  {
    float4 fb[4];
#pragma unroll
    for (int i = 0; i < 4; ++i)
      fb[i] = ((const float4*)(x + (size_t)(row0 + ar + i * 16) * DIM))[ac4];
#pragma unroll
    for (int i = 0; i < 4; ++i) {
      _Float16* dst = &As[0][ar + i * 16][ac4 * 4];
      dst[0] = (_Float16)fb[i].x; dst[1] = (_Float16)fb[i].y;
      dst[2] = (_Float16)fb[i].z; dst[3] = (_Float16)fb[i].w;
    }
  }
#if USE_TDM
  if (wave == 0) __builtin_amdgcn_s_wait_tensorcnt(0);
#endif
  __syncthreads();

  v8f acc[2][2] = {};
  for (int c = 0; c < NCHUNK; ++c) {
    const int cur = c & 1, nxt = cur ^ 1;
    const bool more = (c + 1 < NCHUNK);
    const int kc1 = (c + 1) * BK;
    float4 fb[4];
    if (more) {
#if USE_TDM
      if (wave == 0)
        tdm_load_2d(lds_off(&Bs[nxt][0][0]), wT + (size_t)col0 * DIM + kc1,
                    BK, BNQ, DIM, NQKV, DIM, 4, 3, 1);
#else
#pragma unroll
      for (int i = 0; i < 4; ++i) {
        const int u = tid + i * 256;
        const int n = u >> 3, c8 = u & 7;
        *((uint4*)&Bs[nxt][n][c8 * 8]) =
            ((const uint4*)(wT + (size_t)(col0 + n) * DIM + kc1))[c8];
      }
#endif
#pragma unroll
      for (int i = 0; i < 4; ++i)
        fb[i] =
            ((const float4*)(x + (size_t)(row0 + ar + i * 16) * DIM + kc1))[ac4];
    }
    // ---- compute on current buffer (overlaps outstanding loads) ----
#pragma unroll
    for (int ks = 0; ks < BK; ks += 32) {
      const v16h a0 = load_frag_A(&As[cur][0][0], LDST, wm, ks, lane);
      const v16h a1 = load_frag_A(&As[cur][0][0], LDST, wm + 16, ks, lane);
      const v16h b0 = load_frag_B(&Bs[cur][0][0], LDST, wn, ks, lane);
      const v16h b1 = load_frag_B(&Bs[cur][0][0], LDST, wn + 16, ks, lane);
      acc[0][0] = wmma_f16(a0, b0, acc[0][0]);
      acc[0][1] = wmma_f16(a0, b1, acc[0][1]);
      acc[1][0] = wmma_f16(a1, b0, acc[1][0]);
      acc[1][1] = wmma_f16(a1, b1, acc[1][1]);
    }
    if (more) {
#pragma unroll
      for (int i = 0; i < 4; ++i) {
        _Float16* dst = &As[nxt][ar + i * 16][ac4 * 4];
        dst[0] = (_Float16)fb[i].x; dst[1] = (_Float16)fb[i].y;
        dst[2] = (_Float16)fb[i].z; dst[3] = (_Float16)fb[i].w;
      }
#if USE_TDM
      if (wave == 0) __builtin_amdgcn_s_wait_tensorcnt(0);
#endif
    }
    __syncthreads();
  }

  // Epilogue: +bias, scale q, scatter f16 into [b][h][tok][d]
  const float qscale = 0.17677669529663687f;  // 32^-0.5
  const int nloc = lane & 15, hi = lane >> 4;
  const int bbase = row0 / NTOK;              // block-uniform
  const int tbase = row0 % NTOK;
#pragma unroll
  for (int i = 0; i < 2; ++i) {
#pragma unroll
    for (int j = 0; j < 2; ++j) {
#pragma unroll
      for (int v = 0; v < 8; ++v) {
        const int m = wm + i * 16 + v + hi * 8;
        const int c = col0 + wn + j * 16 + nloc;
        float val = acc[i][j][v] + qkv_b[c];
        const int which = c / DIM;            // 0=q 1=k 2=v
        const int h = (c % DIM) >> 5;
        const int d = c & 31;
        if (which == 0) val *= qscale;
        int tok = tbase + m, b = bbase;       // avoid per-elem div/mod by 49
        if (tok >= 2 * NTOK)      { b += 2; tok -= 2 * NTOK; }
        else if (tok >= NTOK)     { b += 1; tok -= NTOK; }
        const size_t o =
            (((size_t)b * NUM_HEADS + h) * NTOK + tok) * HEAD_DIM + d;
        _Float16* dst = (which == 0) ? qws : (which == 1) ? kws : vws;
        dst[o] = (_Float16)val;
      }
    }
  }
}

// ---------------------------------------------------------------------------
// Kernel 2: attention per (b, head). TDM stages q/k with hardware zero-pad
// (tensor_dim0 = 49*32 < tile_dim0 = 64*32) and padded LDS rows; v staged
// manually transposed (overlaps the TDM). S = q k^T (+bias), softmax, O = P v.
// ---------------------------------------------------------------------------
__global__ __launch_bounds__(128) void attn_kernel(
    const _Float16* __restrict__ qws, const _Float16* __restrict__ kws,
    const _Float16* __restrict__ vws, const float* __restrict__ biasp,
    _Float16* __restrict__ ows) {
  __shared__ _Float16 qs[64][HEAD_DIM + LDPAD];
  __shared__ _Float16 ks[64][HEAD_DIM + LDPAD];
  __shared__ _Float16 vT[HEAD_DIM][64 + LDPAD];   // [d][tok]
  __shared__ _Float16 Ps[64][64 + LDPAD];

  const int bh = blockIdx.x;
  const int b = bh / NUM_HEADS, h = bh % NUM_HEADS;
  const size_t base = (size_t)bh * NTOK * HEAD_DIM;
  const int tid = threadIdx.x, wave = tid >> 5, lane = tid & 31;

#if USE_TDM
  if (wave == 0) {
    // one row of 2048 f16; elements >= 1568 are OOB -> zero (token padding);
    // pad 4 dwords after every 16 dwords (32 f16) -> qs[64][40] layout.
    tdm_load_2d(lds_off(&qs[0][0]), qws + base, 64 * HEAD_DIM, 1,
                NTOK * HEAD_DIM, 1, 64 * HEAD_DIM, 3, 3, 1);
    tdm_load_2d(lds_off(&ks[0][0]), kws + base, 64 * HEAD_DIM, 1,
                NTOK * HEAD_DIM, 1, 64 * HEAD_DIM, 3, 3, 1);
  }
  for (int i = tid; i < 64 * HEAD_DIM; i += 128) {
    const int tok = i >> 5, d = i & 31;
    _Float16 vv = (_Float16)0.f;
    if (tok < NTOK) vv = vws[base + i];
    vT[d][tok] = vv;
  }
  if (wave == 0) __builtin_amdgcn_s_wait_tensorcnt(0);
#else
  for (int i = tid; i < 64 * HEAD_DIM; i += 128) {
    const int tok = i >> 5, d = i & 31;
    _Float16 qv = (_Float16)0.f, kv = (_Float16)0.f, vv = (_Float16)0.f;
    if (tok < NTOK) { qv = qws[base + i]; kv = kws[base + i]; vv = vws[base + i]; }
    qs[tok][d] = qv; ks[tok][d] = kv; vT[d][tok] = vv;
  }
#endif
  __syncthreads();

  const int m0 = wave * 16;
  const int nloc = lane & 15, hi = lane >> 4;

  // S = q k^T : K = 32 in one WMMA per 16x16 tile
  const v16h aq = load_frag_A(&qs[0][0], HEAD_DIM + LDPAD, m0, 0, lane);
  v8f s[4];
#pragma unroll
  for (int nt = 0; nt < 4; ++nt) {
    const v16h bk = load_frag_B(&ks[0][0], HEAD_DIM + LDPAD, nt * 16, 0, lane);
    v8f z = {};
    s[nt] = wmma_f16(aq, bk, z);
  }

  // bias + row softmax (row lives in one half-wave's 16 lanes x 4 tiles)
  const float* hb = biasp + (size_t)h * 64 * 64;
#pragma unroll
  for (int v = 0; v < 8; ++v) {
    const int m = m0 + v + hi * 8;
    float r[4];
    float mx = -1e30f;
#pragma unroll
    for (int nt = 0; nt < 4; ++nt) {
      r[nt] = s[nt][v] + hb[m * 64 + nt * 16 + nloc];
      mx = fmaxf(mx, r[nt]);
    }
#pragma unroll
    for (int off = 8; off >= 1; off >>= 1)
      mx = fmaxf(mx, __shfl_xor(mx, off, 32));
    float sum = 0.f;
#pragma unroll
    for (int nt = 0; nt < 4; ++nt) {
      r[nt] = __expf(r[nt] - mx);
      sum += r[nt];
    }
#pragma unroll
    for (int off = 8; off >= 1; off >>= 1)
      sum += __shfl_xor(sum, off, 32);
    const float inv = 1.0f / (sum + 1e-30f);
#pragma unroll
    for (int nt = 0; nt < 4; ++nt)
      Ps[m][nt * 16 + nloc] = (_Float16)(r[nt] * inv);
  }
  __syncthreads();

  // O = P v : K = 64 (2 steps), N = 32 (2 tiles)
  v8f o0 = {}, o1 = {};
#pragma unroll
  for (int kt = 0; kt < 2; ++kt) {
    const v16h pa = load_frag_A(&Ps[0][0], 64 + LDPAD, m0, kt * 32, lane);
    const v16h v0 = load_frag_B(&vT[0][0], 64 + LDPAD, 0, kt * 32, lane);
    const v16h v1 = load_frag_B(&vT[0][0], 64 + LDPAD, 16, kt * 32, lane);
    o0 = wmma_f16(pa, v0, o0);
    o1 = wmma_f16(pa, v1, o1);
  }

#pragma unroll
  for (int v = 0; v < 8; ++v) {
    const int m = m0 + v + hi * 8;
    if (m < NTOK) {
      const size_t ro = ((size_t)b * NTOK + m) * DIM + h * HEAD_DIM;
      ows[ro + nloc]      = (_Float16)o0[v];
      ows[ro + 16 + nloc] = (_Float16)o1[v];
    }
  }
}

// ---------------------------------------------------------------------------
// Kernel 3: proj GEMM. [200704x384](f16) x [384x384](f16) + bias -> f32.
// Both tiles via TDM, double-buffered pipeline as in kernel 1.
// ---------------------------------------------------------------------------
__global__ __launch_bounds__(256) void proj_gemm_kernel(
    const _Float16* __restrict__ ain, const _Float16* __restrict__ wT,
    const float* __restrict__ proj_b, float* __restrict__ out) {
  __shared__ _Float16 As[2][BM][LDST];
  __shared__ _Float16 Bs[2][BNQ][LDST];
  const int tiles_n = DIM / BNQ;          // 3
  const int tn = blockIdx.x % tiles_n;
  const int tm = blockIdx.x / tiles_n;
  const int row0 = tm * BM, col0 = tn * BNQ;
  const int tid  = threadIdx.x;
  const int wave = tid >> 5, lane = tid & 31;
  const int wm = (wave & 1) * 32;
  const int wn = (wave >> 1) * 32;

#if USE_TDM
  if (wave == 0) {
    tdm_load_2d(lds_off(&As[0][0][0]), ain + (size_t)row0 * DIM,
                BK, BM, DIM, NROWS, DIM, 4, 3, 1);
    tdm_load_2d(lds_off(&Bs[0][0][0]), wT + (size_t)col0 * DIM,
                BK, BNQ, DIM, DIM, DIM, 4, 3, 1);
    __builtin_amdgcn_s_wait_tensorcnt(0);
  }
#else
#pragma unroll
  for (int i = 0; i < 2; ++i) {
    const int u = tid + i * 256;
    const int r = u >> 3, c8 = u & 7;
    *((uint4*)&As[0][r][c8 * 8]) =
        ((const uint4*)(ain + (size_t)(row0 + r) * DIM))[c8];
  }
#pragma unroll
  for (int i = 0; i < 4; ++i) {
    const int u = tid + i * 256;
    const int n = u >> 3, c8 = u & 7;
    *((uint4*)&Bs[0][n][c8 * 8]) =
        ((const uint4*)(wT + (size_t)(col0 + n) * DIM))[c8];
  }
#endif
  __syncthreads();

  v8f acc[2][2] = {};
  for (int c = 0; c < NCHUNK; ++c) {
    const int cur = c & 1, nxt = cur ^ 1;
    const bool more = (c + 1 < NCHUNK);
    const int kc1 = (c + 1) * BK;
    if (more) {
#if USE_TDM
      if (wave == 0) {
        tdm_load_2d(lds_off(&As[nxt][0][0]), ain + (size_t)row0 * DIM + kc1,
                    BK, BM, DIM, NROWS, DIM, 4, 3, 1);
        tdm_load_2d(lds_off(&Bs[nxt][0][0]), wT + (size_t)col0 * DIM + kc1,
                    BK, BNQ, DIM, DIM, DIM, 4, 3, 1);
      }
#else
#pragma unroll
      for (int i = 0; i < 2; ++i) {
        const int u = tid + i * 256;
        const int r = u >> 3, c8 = u & 7;
        *((uint4*)&As[nxt][r][c8 * 8]) =
            ((const uint4*)(ain + (size_t)(row0 + r) * DIM + kc1))[c8];
      }
#pragma unroll
      for (int i = 0; i < 4; ++i) {
        const int u = tid + i * 256;
        const int n = u >> 3, c8 = u & 7;
        *((uint4*)&Bs[nxt][n][c8 * 8]) =
            ((const uint4*)(wT + (size_t)(col0 + n) * DIM + kc1))[c8];
      }
#endif
    }
#pragma unroll
    for (int ks = 0; ks < BK; ks += 32) {
      const v16h a0 = load_frag_A(&As[cur][0][0], LDST, wm, ks, lane);
      const v16h a1 = load_frag_A(&As[cur][0][0], LDST, wm + 16, ks, lane);
      const v16h b0 = load_frag_B(&Bs[cur][0][0], LDST, wn, ks, lane);
      const v16h b1 = load_frag_B(&Bs[cur][0][0], LDST, wn + 16, ks, lane);
      acc[0][0] = wmma_f16(a0, b0, acc[0][0]);
      acc[0][1] = wmma_f16(a0, b1, acc[0][1]);
      acc[1][0] = wmma_f16(a1, b0, acc[1][0]);
      acc[1][1] = wmma_f16(a1, b1, acc[1][1]);
    }
#if USE_TDM
    if (more && wave == 0) __builtin_amdgcn_s_wait_tensorcnt(0);
#endif
    __syncthreads();
  }

  const int nloc = lane & 15, hi = lane >> 4;
#pragma unroll
  for (int i = 0; i < 2; ++i) {
#pragma unroll
    for (int j = 0; j < 2; ++j) {
#pragma unroll
      for (int v = 0; v < 8; ++v) {
        const int m = wm + i * 16 + v + hi * 8;
        const int c = col0 + wn + j * 16 + nloc;
        out[(size_t)(row0 + m) * DIM + c] = acc[i][j][v] + proj_b[c];
      }
    }
  }
}

// ---------------------------------------------------------------------------
// Host launcher. Workspace (bytes): qkvwT@0 (884736) | projwT@884736 (294912) |
// biasp@1179648 (196608) | q/k/v f16 @1376256 (3x154140672) | o f16 (154140672)
// Total ~618 MB.
// ---------------------------------------------------------------------------
extern "C" void kernel_launch(void* const* d_in, const int* in_sizes, int n_in,
                              void* d_out, int out_size, void* d_ws, size_t ws_size,
                              hipStream_t stream) {
  const float* x          = (const float*)d_in[0];
  const float* qkv_w      = (const float*)d_in[1];
  const float* qkv_b      = (const float*)d_in[2];
  const float* proj_w     = (const float*)d_in[3];
  const float* proj_b     = (const float*)d_in[4];
  const float* bias_table = (const float*)d_in[5];
  const int*   rel        = (const int*)d_in[6];
  float* out = (float*)d_out;

  char* ws = (char*)d_ws;
  _Float16* qkvwT  = (_Float16*)(ws + 0);
  _Float16* projwT = (_Float16*)(ws + 884736);
  float*    biasp  = (float*)   (ws + 1179648);
  _Float16* qws    = (_Float16*)(ws + 1376256);
  const size_t QKV_ELEMS = (size_t)NBATCH * NUM_HEADS * NTOK * HEAD_DIM;
  _Float16* kws = qws + QKV_ELEMS;
  _Float16* vws = kws + QKV_ELEMS;
  _Float16* ows = vws + QKV_ELEMS;

  prep_kernel<<<(638976 + 255) / 256, 256, 0, stream>>>(
      qkv_w, proj_w, bias_table, rel, qkvwT, projwT, biasp);

  // QKV GEMM: (200704/64) x (1152/128) = 3136 x 9 tiles
  qkv_gemm_kernel<<<3136 * 9, 256, 0, stream>>>(x, qkvwT, qkv_b, qws, kws, vws);

  attn_kernel<<<NBATCH * NUM_HEADS, 128, 0, stream>>>(qws, kws, vws, biasp, ows);

  // proj GEMM: 3136 x 3 tiles
  proj_gemm_kernel<<<3136 * 3, 256, 0, stream>>>(ows, projwT, proj_b, out);
}
